// IpexWoqLinear_69526930587939
// MI455X (gfx1250) — matrix operand order repaired and checked by reading, not compiled, not run on hardware
//
#include <hip/hip_runtime.h>

// ---------------------------------------------------------------------------
// WOQ linear: Y[m,n] = s[n] * ( (x @ q^T)[m,n] - zp[n]*rowsum_x[m] ) + bias[n]
// M = 4*2048 = 8192, K = 4096, N = 11008.
// GEMM runs in f16 WMMA (codes 0..255 are exact in f16), f32 accumulate.
// ---------------------------------------------------------------------------

typedef __attribute__((ext_vector_type(16))) _Float16 v16h;
typedef __attribute__((ext_vector_type(8)))  _Float16 v8h;
typedef __attribute__((ext_vector_type(4)))  _Float16 v4h;
typedef __attribute__((ext_vector_type(8)))  float    v8f;
typedef __attribute__((ext_vector_type(4)))  float    f4;
typedef __attribute__((ext_vector_type(4)))  int      i4;

#define MDIM 8192
#define KDIM 4096
#define NDIM 11008

#define BM 128
#define BN 128
#define BK 64
#define LDK 80   // padded LDS row stride in halves (160 B, keeps 16B alignment)

__global__ __launch_bounds__(256)
void woq_gemm_wmma(const float* __restrict__ x,
                   const int*   __restrict__ q,
                   const float* __restrict__ scales,
                   const float* __restrict__ zps,
                   const float* __restrict__ bias,
                   float*       __restrict__ out)
{
    __shared__ _Float16 As[BM * LDK];       // x tile, f16, [m][k]
    __shared__ _Float16 Bs[BN * LDK];       // q tile, f16, [n][k]
    __shared__ float    psum[16 * BM];      // per-colIdx partial row sums
    __shared__ float    rowsum[BM];         // final sum_k x[m,k] for this block

    const int tid   = threadIdx.x;
    const int lane  = tid & 31;
    const int wid   = tid >> 5;             // 0..7
    const int waveM = wid & 1;              // 2 wave-rows  (64 M each)
    const int waveN = wid >> 1;             // 4 wave-cols  (32 N each)

    const int nBlock = blockIdx.x;          // 0..85
    const int mBlock = blockIdx.y;          // 0..63

    // tile-loader mapping: 16 k-chunks of 4 elements x 16 row groups of 8 rows
    const int colIdx  = tid & 15;           // 4-element chunk along K
    const int rowBase = (tid >> 4) * 8;     // first of 8 rows handled

    const float* xg = x + (size_t)(mBlock * BM) * KDIM;
    const int*   qg = q + (size_t)(nBlock * BN) * KDIM;

    float partial[8];
#pragma unroll
    for (int r = 0; r < 8; ++r) partial[r] = 0.0f;

    v8f acc[4][2];
#pragma unroll
    for (int mt = 0; mt < 4; ++mt)
#pragma unroll
        for (int nt = 0; nt < 2; ++nt)
            acc[mt][nt] = (v8f)(0.0f);

    for (int kt = 0; kt < KDIM / BK; ++kt) {
        const int k0 = kt * BK;

        // ---- stage K-tile: global -> registers (coalesced 16B loads) ----
        f4 av[8];
        i4 bv[8];
#pragma unroll
        for (int r = 0; r < 8; ++r) {
            av[r] = *(const f4*)(xg + (size_t)(rowBase + r) * KDIM + k0 + colIdx * 4);
            bv[r] = *(const i4*)(qg + (size_t)(rowBase + r) * KDIM + k0 + colIdx * 4);
        }
        if (kt + 1 < KDIM / BK) {
            __builtin_prefetch(xg + (size_t)rowBase * KDIM + k0 + BK + colIdx * 4, 0, 1);
            __builtin_prefetch(qg + (size_t)rowBase * KDIM + k0 + BK + colIdx * 4, 0, 1);
        }

        __syncthreads();   // previous iteration's compute is done with LDS

        // ---- convert + store to LDS; fold x row-sum accumulation in ----
#pragma unroll
        for (int r = 0; r < 8; ++r) {
            f4 a = av[r];
            partial[r] += a[0] + a[1] + a[2] + a[3];
            v4h ah = { (_Float16)a[0], (_Float16)a[1], (_Float16)a[2], (_Float16)a[3] };
            *(v4h*)&As[(rowBase + r) * LDK + colIdx * 4] = ah;

            i4 b = bv[r];
            v4h bh = { (_Float16)(float)b[0], (_Float16)(float)b[1],
                       (_Float16)(float)b[2], (_Float16)(float)b[3] };
            *(v4h*)&Bs[(rowBase + r) * LDK + colIdx * 4] = bh;
        }

        __syncthreads();   // tile visible to all waves

        // ---- WMMA compute: 2 K-slices x (4 M-tiles x 2 N-tiles) ----
        const int arow = lane & 15;
        const int kb   = (lane >> 4) * 16;   // A: elems 0-15 = K kb..kb+15
        const int kh   = (lane >> 4) * 16;   // B: elems 0-15 = K kh..kh+15
#pragma unroll
        for (int ks = 0; ks < BK; ks += 32) {
            v16h af[4];
#pragma unroll
            for (int mt = 0; mt < 4; ++mt) {
                const _Float16* p =
                    &As[(waveM * 64 + mt * 16 + arow) * LDK + ks + kb];
                v8h lo = *(const v8h*)__builtin_assume_aligned(p, 16);
                v8h hi = *(const v8h*)__builtin_assume_aligned(p + 8, 16);
                af[mt] = __builtin_shufflevector(lo, hi,
                          0, 1, 2, 3, 4, 5, 6, 7, 8, 9, 10, 11, 12, 13, 14, 15);
            }
            v16h bf[2];
#pragma unroll
            for (int nt = 0; nt < 2; ++nt) {
                const _Float16* p =
                    &Bs[(waveN * 32 + nt * 16 + arow) * LDK + ks + kh];
                bf[nt] = *(const v16h*)__builtin_assume_aligned(p, 16);
            }
#pragma unroll
            for (int mt = 0; mt < 4; ++mt)
#pragma unroll
                for (int nt = 0; nt < 2; ++nt)
                    acc[mt][nt] = __builtin_amdgcn_wmma_f32_16x16x32_f16(
                        /*neg_a=*/false, af[mt],
                        /*neg_b=*/false, bf[nt],
                        /*c_mod=*/(short)0, acc[mt][nt],
                        /*reuse_a=*/false, /*reuse_b=*/false);
        }
    }

    // ---- deterministic row-sum reduction (fixed summation order) ----
#pragma unroll
    for (int r = 0; r < 8; ++r)
        psum[colIdx * BM + rowBase + r] = partial[r];
    __syncthreads();
    if (tid < BM) {
        float s = 0.0f;
#pragma unroll
        for (int c = 0; c < 16; ++c)
            s += psum[c * BM + tid];
        rowsum[tid] = s;
    }
    __syncthreads();

    // ---- epilogue: dequant fold + bias, write C tile ----
    // C layout: lane 0-15 -> n=lane, M=vgpr j; lane 16-31 -> n=lane-16, M=8+j
    const int ncol = lane & 15;
    const int mo   = (lane >> 4) * 8;
#pragma unroll
    for (int nt = 0; nt < 2; ++nt) {
        const int n = nBlock * BN + waveN * 32 + nt * 16 + ncol;
        const float s  = scales[n];
        const float z  = zps[n];
        const float bb = bias[n];
#pragma unroll
        for (int mt = 0; mt < 4; ++mt) {
#pragma unroll
            for (int j = 0; j < 8; ++j) {
                const int lm = waveM * 64 + mt * 16 + mo + j;
                const float v = acc[mt][nt][j];
                out[(size_t)(mBlock * BM + lm) * NDIM + n] =
                    s * (v - z * rowsum[lm]) + bb;
            }
        }
    }
}

extern "C" void kernel_launch(void* const* d_in, const int* in_sizes, int n_in,
                              void* d_out, int out_size, void* d_ws, size_t ws_size,
                              hipStream_t stream) {
    const float* x       = (const float*)d_in[0];
    const int*   qweight = (const int*)d_in[1];
    const float* scales  = (const float*)d_in[2];
    const float* zps     = (const float*)d_in[3];
    const float* bias    = (const float*)d_in[4];
    float*       out     = (float*)d_out;

    dim3 grid(NDIM / BN, MDIM / BM);   // 86 x 64 blocks
    dim3 block(256);                   // 8 wave32s
    woq_gemm_wmma<<<grid, block, 0, stream>>>(x, qweight, scales, zps, bias, out);
}
